// MultiHead_Self_Attention_72756745994584
// MI455X (gfx1250) — compile-verified, matrix operand order checked by reading, not run on hardware
//
#include <hip/hip_runtime.h>
#include <hip/hip_bf16.h>

// ---------------------------------------------------------------------------
// MultiHead self-attention with RoPE, CDNA5 (gfx1250), wave32 + WMMA bf16.
//   B=2, S=2048, D_MODEL=1024, H=16, D_K=64, THETA=10000
// Pipeline:
//   1) proj_qkv_rope : Q/K/V = x @ W^T  (bf16 WMMA, f32 acc), RoPE fused on
//                      Q (pre-scaled 1/8) and K; V stored transposed [B,H,d,S]
//   2) flash_attn    : causal online-softmax attention; K/V tiles staged in
//                      double-buffered LDS via global_load_async_to_lds_b128
//   3) proj_out      : out = attn @ Wo^T, fp32 epilogue (async bf16 A staging)
// ---------------------------------------------------------------------------

#define D_MODEL 1024
#define NUM_HEADS 16
#define D_K 64
#define SEQ 2048
#define BS 4096              // B*S
#define LN_THETA_OVER_32 0.28782313662425572f  // ln(10000)/32

typedef __bf16 bf16_t;
typedef __bf16 v16bf __attribute__((ext_vector_type(16)));
typedef __bf16 v8bf  __attribute__((ext_vector_type(8)));
typedef float  v8f   __attribute__((ext_vector_type(8)));

union U4BF8 { uint4 u; v8bf b; };

__device__ __forceinline__ v8bf ld_bf8(const bf16_t* p) {
    U4BF8 t; t.u = *reinterpret_cast<const uint4*>(p); return t.b;
}
__device__ __forceinline__ v16bf cat8(v8bf lo, v8bf hi) {
    return __builtin_shufflevector(lo, hi, 0,1,2,3,4,5,6,7,8,9,10,11,12,13,14,15);
}
__device__ __forceinline__ v8f wmma_bf16(v16bf a, v16bf b, v8f c) {
    return __builtin_amdgcn_wmma_f32_16x16x32_bf16(false, a, false, b, (short)0, c, false, false);
}

// Async global -> LDS 16B copy (CDNA5 TDM-adjacent path, ASYNCcnt-tracked).
// LDS operand: low 32 bits of the generic pointer == LDS byte offset (ISA 10.2).
__device__ __forceinline__ void async_copy_b128(const void* gsrc, void* ldst) {
    unsigned lds = (unsigned)(unsigned long long)ldst;
    unsigned long long ga = (unsigned long long)gsrc;
    asm volatile("global_load_async_to_lds_b128 %0, %1, off"
                 :: "v"(lds), "v"(ga) : "memory");
}
__device__ __forceinline__ void wait_async0() {
    asm volatile("s_wait_asynccnt 0x0" ::: "memory");
}

// ---------------------------------------------------------------------------
// Kernel 1: fused QKV projection + RoPE.
//   C[m, o] = sum_i X[m, i] * W[o, i]     (M=4096, N=1024, K=1024)
//   grid = (N/128, M/128, 3)  z: 0=Q (rope, *0.125) 1=K (rope) 2=V (transposed)
// ---------------------------------------------------------------------------
__global__ __launch_bounds__(256) void proj_qkv_rope_kernel(
    const float* __restrict__ x,
    const float* __restrict__ Wq, const float* __restrict__ Wk,
    const float* __restrict__ Wv,
    bf16_t* __restrict__ q_ws, bf16_t* __restrict__ k_ws,
    bf16_t* __restrict__ vt_ws)
{
    __shared__ bf16_t As[128 * 40];   // 128 x 32 bf16, pitch 40 (80B rows, 16B aligned chunks)
    __shared__ bf16_t Bsh[128 * 40];

    const int kind  = blockIdx.z;
    const float* W  = (kind == 0) ? Wq : ((kind == 1) ? Wk : Wv);
    const int mbase = blockIdx.y * 128;
    const int nbase = blockIdx.x * 128;
    const int tid   = threadIdx.x;
    const int lane  = tid & 31;
    const int w     = tid >> 5;          // 8 waves: 4 (M) x 2 (N)
    const int wm    = (w & 3) * 32;
    const int wn    = (w >> 2) * 64;
    const int r     = lane & 15;
    const int half  = lane >> 4;

    const v8f zero8 = {0.f,0.f,0.f,0.f,0.f,0.f,0.f,0.f};
    v8f acc[2][4];
#pragma unroll
    for (int mt = 0; mt < 2; ++mt)
#pragma unroll
        for (int nt = 0; nt < 4; ++nt) acc[mt][nt] = zero8;

    const int srow = tid >> 3;           // staging row (0..31), 8 threads/row
    const int skc  = tid & 7;            // staging chunk (float4) within row

    for (int k0 = 0; k0 < 1024; k0 += 32) {
        __syncthreads();
        // batched loads: 8 b128 loads in flight before any conversion
        float4 va[4], vb[4];
#pragma unroll
        for (int i = 0; i < 4; ++i) {
            int row = srow + i * 32;     // tid>>3 + i*32 covers 0..127
            va[i] = *reinterpret_cast<const float4*>(
                &x[(mbase + row) * 1024 + k0 + skc * 4]);
            vb[i] = *reinterpret_cast<const float4*>(
                &W[(nbase + row) * 1024 + k0 + skc * 4]);
        }
#pragma unroll
        for (int i = 0; i < 4; ++i) {
            int row = srow + i * 32;
            bf16_t* da = &As[row * 40 + skc * 4];
            da[0] = (bf16_t)va[i].x; da[1] = (bf16_t)va[i].y;
            da[2] = (bf16_t)va[i].z; da[3] = (bf16_t)va[i].w;
            bf16_t* db = &Bsh[row * 40 + skc * 4];
            db[0] = (bf16_t)vb[i].x; db[1] = (bf16_t)vb[i].y;
            db[2] = (bf16_t)vb[i].z; db[3] = (bf16_t)vb[i].w;
        }
        __syncthreads();

        if (k0 + 32 < 1024) {   // prefetch next K-tile into caches
            __builtin_prefetch(&x[(mbase + (tid >> 1)) * 1024 + k0 + 32 + (tid & 1) * 16], 0, 1);
            __builtin_prefetch(&W[(nbase + (tid >> 1)) * 1024 + k0 + 32 + (tid & 1) * 16], 0, 1);
        }

        v16bf a[2], b[4];
#pragma unroll
        for (int mt = 0; mt < 2; ++mt) {
            const bf16_t* base = &As[(wm + mt * 16 + r) * 40];
            a[mt] = cat8(ld_bf8(base + half * 8), ld_bf8(base + 16 + half * 8));
        }
#pragma unroll
        for (int nt = 0; nt < 4; ++nt) {
            const bf16_t* base = &Bsh[(wn + nt * 16 + r) * 40];
            b[nt] = cat8(ld_bf8(base + half * 8), ld_bf8(base + 16 + half * 8));
        }
#pragma unroll
        for (int mt = 0; mt < 2; ++mt)
#pragma unroll
            for (int nt = 0; nt < 4; ++nt)
                acc[mt][nt] = wmma_bf16(a[mt], b[nt], acc[mt][nt]);
    }

    // epilogue: RoPE (Q/K) or transposed store (V)
#pragma unroll
    for (int mt = 0; mt < 2; ++mt) {
#pragma unroll
        for (int nt = 0; nt < 4; ++nt) {
            const int o = nbase + wn + nt * 16 + r;   // output channel (per lane)
            const int h = o >> 6;
            const int d = o & 63;
            const float inv = __expf(-(float)(d >> 1) * LN_THETA_OVER_32);
#pragma unroll
            for (int v = 0; v < 8; ++v) {
                const int m = mbase + wm + mt * 16 + v + 8 * half;
                const int s = m & (SEQ - 1);
                const int bb = m >> 11;
                const float val = acc[mt][nt][v];
                const float partner = __shfl_xor(val, 1);   // RoPE pair is adjacent lane
                if (kind == 2) {
                    vt_ws[((bb * NUM_HEADS + h) * D_K + d) * SEQ + s] = (bf16_t)val;
                } else {
                    float sn, cs;
                    __sincosf((float)s * inv, &sn, &cs);
                    float rot = val * cs + ((d & 1) ? partner : -partner) * sn;
                    if (kind == 0) rot *= 0.125f;           // 1/sqrt(D_K) folded into Q
                    bf16_t* dst = (kind == 0) ? q_ws : k_ws;
                    dst[((bb * NUM_HEADS + h) * SEQ + s) * D_K + d] = (bf16_t)rot;
                }
            }
        }
    }
}

// ---------------------------------------------------------------------------
// Kernel 2: causal flash attention.
//   grid = (S/128, B*H), block 256; wave w owns q rows [qbase+16w, +16).
//   K / V^T tiles double-buffered in LDS via async global->LDS copies,
//   next tile's copy overlapped with current tile's WMMA + softmax.
// ---------------------------------------------------------------------------
__global__ __launch_bounds__(256) void flash_attn_kernel(
    const bf16_t* __restrict__ q_ws, const bf16_t* __restrict__ k_ws,
    const bf16_t* __restrict__ vt_ws, bf16_t* __restrict__ attn_ws)
{
    __shared__ bf16_t Kls[2][64 * 72];   // 64 keys x 64 d, pitch 72
    __shared__ bf16_t Vls[2][64 * 72];   // 64 d x 64 keys, pitch 72
    __shared__ bf16_t Pls[8][16 * 72];   // per-wave P staging (C-layout -> A-layout)

    const int tid  = threadIdx.x;
    const int lane = tid & 31;
    const int w    = tid >> 5;
    const int r    = lane & 15;
    const int half = lane >> 4;
    const int bh   = blockIdx.y;
    const int b    = bh >> 4;
    const int h    = bh & 15;
    const int qbase  = blockIdx.x * 128;
    const int qrow_w = qbase + w * 16;

    const bf16_t* Qp = q_ws  + (size_t)(b * NUM_HEADS + h) * SEQ * D_K;
    const bf16_t* Kp = k_ws  + (size_t)(b * NUM_HEADS + h) * SEQ * D_K;
    const bf16_t* Vt = vt_ws + (size_t)(b * NUM_HEADS + h) * D_K * SEQ;

    // Q A-fragments for this wave's 16 rows (d = 0..63 -> two 16x32 frags)
    v16bf qa[2];
    {
        const bf16_t* qr = Qp + (qrow_w + r) * D_K;
        qa[0] = cat8(ld_bf8(qr + half * 8),      ld_bf8(qr + 16 + half * 8));
        qa[1] = cat8(ld_bf8(qr + 32 + half * 8), ld_bf8(qr + 48 + half * 8));
    }

    const v8f zero8 = {0.f,0.f,0.f,0.f,0.f,0.f,0.f,0.f};
    v8f accO[4];
#pragma unroll
    for (int nt = 0; nt < 4; ++nt) accO[nt] = zero8;
    float mrow[8], lrow[8];
#pragma unroll
    for (int v = 0; v < 8; ++v) { mrow[v] = -1e30f; lrow[v] = 0.f; }

    // staging: 64 rows x 8 chunks(16B) per tile; 2 chunks/thread/matrix
    const int c_row = tid >> 3;          // uses tid, +32 for second chunk
    const int c_cc  = tid & 7;

    const int nkb = (qbase >> 6) + 2;    // causal extent of this q-tile

    // stage tile 0 into buffer 0
#pragma unroll
    for (int i = 0; i < 2; ++i) {
        int row = c_row + i * 32;
        async_copy_b128(Kp + row * D_K + c_cc * 8,       &Kls[0][row * 72 + c_cc * 8]);
        async_copy_b128(Vt + row * SEQ + c_cc * 8,       &Vls[0][row * 72 + c_cc * 8]);
    }

    for (int kb = 0; kb < nkb; ++kb) {
        const int kbase = kb * 64;
        const int cur = kb & 1;

        wait_async0();          // this wave's staged chunks arrived
        __syncthreads();        // everyone's chunks arrived; prior reads done

        if (kb + 1 < nkb) {     // overlap next tile's async copy with compute
            const int nb = (kb + 1) & 1;
            const int nbas = kbase + 64;
#pragma unroll
            for (int i = 0; i < 2; ++i) {
                int row = c_row + i * 32;
                async_copy_b128(Kp + (nbas + row) * D_K + c_cc * 8,
                                &Kls[nb][row * 72 + c_cc * 8]);
                async_copy_b128(Vt + row * SEQ + nbas + c_cc * 8,
                                &Vls[nb][row * 72 + c_cc * 8]);
            }
        }

        // S = Q K^T for 64 keys (K frags from LDS)
        v8f s[4];
#pragma unroll
        for (int nt = 0; nt < 4; ++nt) s[nt] = zero8;
#pragma unroll
        for (int nt = 0; nt < 4; ++nt) {
            const bf16_t* kr = &Kls[cur][(nt * 16 + r) * 72];
            v16bf bk0 = cat8(ld_bf8(kr + half * 8),      ld_bf8(kr + 16 + half * 8));
            v16bf bk1 = cat8(ld_bf8(kr + 32 + half * 8), ld_bf8(kr + 48 + half * 8));
            s[nt] = wmma_bf16(qa[0], bk0, s[nt]);
            s[nt] = wmma_bf16(qa[1], bk1, s[nt]);
        }

        const bool need_mask = (kbase + 63) > qrow_w;   // wave-uniform

        // online softmax, stats per row (replicated across 16-lane groups)
        float scale_v[8];
#pragma unroll
        for (int v = 0; v < 8; ++v) {
            const int qr_g = qrow_w + v + 8 * half;
            float mx = -1e30f;
#pragma unroll
            for (int nt = 0; nt < 4; ++nt) {
                float val = s[nt][v];
                if (need_mask && (kbase + nt * 16 + r) > qr_g) { val = -1e30f; s[nt][v] = val; }
                mx = fmaxf(mx, val);
            }
#pragma unroll
            for (int msk = 1; msk < 16; msk <<= 1) mx = fmaxf(mx, __shfl_xor(mx, msk));
            const float mnew = fmaxf(mrow[v], mx);
            const float sc = __expf(mrow[v] - mnew);
            float rsum = 0.f;
#pragma unroll
            for (int nt = 0; nt < 4; ++nt) {
                float p = __expf(s[nt][v] - mnew);
                s[nt][v] = p;
                rsum += p;
            }
#pragma unroll
            for (int msk = 1; msk < 16; msk <<= 1) rsum += __shfl_xor(rsum, msk);
            lrow[v] = lrow[v] * sc + rsum;
            mrow[v] = mnew;
            scale_v[v] = sc;
        }
#pragma unroll
        for (int nt = 0; nt < 4; ++nt)
#pragma unroll
            for (int v = 0; v < 8; ++v) accO[nt][v] *= scale_v[v];

        // P: C-layout -> LDS -> A-layout (same-wave DS ops are in-order)
        bf16_t* Pb = &Pls[w][0];
#pragma unroll
        for (int nt = 0; nt < 4; ++nt)
#pragma unroll
            for (int v = 0; v < 8; ++v)
                Pb[(v + 8 * half) * 72 + nt * 16 + r] = (bf16_t)s[nt][v];
        __builtin_amdgcn_wave_barrier();

        v16bf pa[2];
        {
            const bf16_t* pr = Pb + r * 72;
            pa[0] = cat8(ld_bf8(pr + half * 8),      ld_bf8(pr + 16 + half * 8));
            pa[1] = cat8(ld_bf8(pr + 32 + half * 8), ld_bf8(pr + 48 + half * 8));
        }

        // O += P V   (V^T frags from LDS; rows contiguous over keys)
#pragma unroll
        for (int nt = 0; nt < 4; ++nt) {
            const bf16_t* vr = &Vls[cur][(nt * 16 + r) * 72];
            v16bf bv0 = cat8(ld_bf8(vr + half * 8),      ld_bf8(vr + 16 + half * 8));
            v16bf bv1 = cat8(ld_bf8(vr + 32 + half * 8), ld_bf8(vr + 48 + half * 8));
            accO[nt] = wmma_bf16(pa[0], bv0, accO[nt]);
            accO[nt] = wmma_bf16(pa[1], bv1, accO[nt]);
        }
    }

    // normalize and store attn output row-major [BS, D_MODEL] bf16
#pragma unroll
    for (int nt = 0; nt < 4; ++nt)
#pragma unroll
        for (int v = 0; v < 8; ++v) {
            const int qr_g = qrow_w + v + 8 * half;
            const float o = accO[nt][v] / lrow[v];
            attn_ws[((size_t)(b * SEQ + qr_g)) * D_MODEL + h * D_K + nt * 16 + r] = (bf16_t)o;
        }
}

// ---------------------------------------------------------------------------
// Kernel 3: output projection out[m, o] = sum_i attn[m, i] * Wo[o, i], fp32 out
//   A tile is already bf16 -> async global->LDS copies; B converted manually.
// ---------------------------------------------------------------------------
__global__ __launch_bounds__(256) void proj_out_kernel(
    const bf16_t* __restrict__ attn, const float* __restrict__ Wo,
    float* __restrict__ out)
{
    __shared__ bf16_t As[128 * 40];
    __shared__ bf16_t Bsh[128 * 40];

    const int mbase = blockIdx.y * 128;
    const int nbase = blockIdx.x * 128;
    const int tid   = threadIdx.x;
    const int lane  = tid & 31;
    const int w     = tid >> 5;
    const int wm    = (w & 3) * 32;
    const int wn    = (w >> 2) * 64;
    const int r     = lane & 15;
    const int half  = lane >> 4;

    const v8f zero8 = {0.f,0.f,0.f,0.f,0.f,0.f,0.f,0.f};
    v8f acc[2][4];
#pragma unroll
    for (int mt = 0; mt < 2; ++mt)
#pragma unroll
        for (int nt = 0; nt < 4; ++nt) acc[mt][nt] = zero8;

    const int arow = tid >> 2;           // A staging: 128 rows x 4 chunks(16B)
    const int akc  = tid & 3;
    const int srow = tid >> 3;           // B staging: 128 rows x 8 float4
    const int skc  = tid & 7;

    for (int k0 = 0; k0 < 1024; k0 += 32) {
        __syncthreads();
        // A (bf16) -> LDS, async 16B copies (2/thread)
#pragma unroll
        for (int i = 0; i < 2; ++i) {
            int row = arow + i * 64;
            async_copy_b128(&attn[(mbase + row) * 1024 + k0 + akc * 8],
                            &As[row * 40 + akc * 8]);
        }
        // B (Wo) fp32 -> bf16, batched loads first
        float4 vb[4];
#pragma unroll
        for (int i = 0; i < 4; ++i) {
            int row = srow + i * 32;
            vb[i] = *reinterpret_cast<const float4*>(
                &Wo[(nbase + row) * 1024 + k0 + skc * 4]);
        }
#pragma unroll
        for (int i = 0; i < 4; ++i) {
            int row = srow + i * 32;
            bf16_t* db = &Bsh[row * 40 + skc * 4];
            db[0] = (bf16_t)vb[i].x; db[1] = (bf16_t)vb[i].y;
            db[2] = (bf16_t)vb[i].z; db[3] = (bf16_t)vb[i].w;
        }
        wait_async0();
        __syncthreads();

        if (k0 + 32 < 1024) {
            __builtin_prefetch(&attn[(mbase + (tid >> 1)) * 1024 + k0 + 32 + (tid & 1) * 16], 0, 1);
            __builtin_prefetch(&Wo[(nbase + (tid >> 1)) * 1024 + k0 + 32 + (tid & 1) * 16], 0, 1);
        }

        v16bf a[2], b[4];
#pragma unroll
        for (int mt = 0; mt < 2; ++mt) {
            const bf16_t* base = &As[(wm + mt * 16 + r) * 40];
            a[mt] = cat8(ld_bf8(base + half * 8), ld_bf8(base + 16 + half * 8));
        }
#pragma unroll
        for (int nt = 0; nt < 4; ++nt) {
            const bf16_t* base = &Bsh[(wn + nt * 16 + r) * 40];
            b[nt] = cat8(ld_bf8(base + half * 8), ld_bf8(base + 16 + half * 8));
        }
#pragma unroll
        for (int mt = 0; mt < 2; ++mt)
#pragma unroll
            for (int nt = 0; nt < 4; ++nt)
                acc[mt][nt] = wmma_bf16(a[mt], b[nt], acc[mt][nt]);
    }

#pragma unroll
    for (int mt = 0; mt < 2; ++mt)
#pragma unroll
        for (int nt = 0; nt < 4; ++nt) {
            const int o = nbase + wn + nt * 16 + r;
#pragma unroll
            for (int v = 0; v < 8; ++v) {
                const int m = mbase + wm + mt * 16 + v + 8 * half;
                out[(size_t)m * D_MODEL + o] = acc[mt][nt][v];
            }
        }
}

// ---------------------------------------------------------------------------
extern "C" void kernel_launch(void* const* d_in, const int* in_sizes, int n_in,
                              void* d_out, int out_size, void* d_ws, size_t ws_size,
                              hipStream_t stream) {
    const float* x  = (const float*)d_in[0];
    // d_in[1] = token_positions (arange(S)) — positions are implicit in s
    const float* Wq = (const float*)d_in[2];
    const float* Wk = (const float*)d_in[3];
    const float* Wv = (const float*)d_in[4];
    const float* Wo = (const float*)d_in[5];

    char* ws = (char*)d_ws;
    const size_t BUF = (size_t)BS * D_MODEL * sizeof(bf16_t);   // 8 MiB each
    bf16_t* q_ws    = (bf16_t*)(ws);
    bf16_t* k_ws    = (bf16_t*)(ws + BUF);
    bf16_t* vt_ws   = (bf16_t*)(ws + 2 * BUF);
    bf16_t* attn_ws = (bf16_t*)(ws + 3 * BUF);

    dim3 blk(256);
    proj_qkv_rope_kernel<<<dim3(8, 32, 3), blk, 0, stream>>>(
        x, Wq, Wk, Wv, q_ws, k_ws, vt_ws);
    flash_attn_kernel<<<dim3(16, 32), blk, 0, stream>>>(
        q_ws, k_ws, vt_ws, attn_ws);
    proj_out_kernel<<<dim3(8, 32), blk, 0, stream>>>(
        attn_ws, Wo, (float*)d_out);
}
